// TripletLoss_39599598469579
// MI455X (gfx1250) — matrix operand order repaired and checked by reading, not compiled, not run on hardware
//
#include <hip/hip_runtime.h>
#include <stdint.h>

// Triplet loss (batch-hard) for MI455X / gfx1250.
// Gram matrix via v_wmma_f32_16x16x32_bf16 with precomputed bf16 hi/lo split
// (hi*hi + hi*lo + lo*hi ~ fp32 accuracy). Column tiles streamed into
// double-buffered LDS with global_load_async_to_lds_b128 (ASYNCcnt),
// mining done on squared distances (no sqrt in the hot loop).

typedef __bf16 bf16_t;
typedef __attribute__((ext_vector_type(16))) __bf16 v16bf;
typedef __attribute__((ext_vector_type(8)))  float  v8f;

#define NROWS 8192
#define DIM   128
#define TL_MARGIN 0.3f

union frag_u { v16bf v; uint4 q[2]; };
union pack8  { bf16_t b[8]; uint4 q; };

static __device__ __forceinline__ uint32_t lds_off(const void* p) {
    // generic pointer to LDS: low 32 bits are the wave-relative LDS address
    return (uint32_t)(uintptr_t)p;
}

static __device__ __forceinline__ void async_copy_b128(uint32_t dst_lds, const void* src) {
    asm volatile("global_load_async_to_lds_b128 %0, %1, off"
                 :: "v"(dst_lds), "v"((uint64_t)(uintptr_t)src)
                 : "memory");
}

static __device__ __forceinline__ void wait_async0() {
#if __has_builtin(__builtin_amdgcn_s_wait_asynccnt)
    __builtin_amdgcn_s_wait_asynccnt(0);
#else
    asm volatile("s_wait_asynccnt 0x0" ::: "memory");
#endif
}

// ---------- Kernel 1: bf16 hi/lo split + row norms + init ----------
__global__ void tl_prep(const float* __restrict__ emb,
                        bf16_t* __restrict__ preHi,
                        bf16_t* __restrict__ preLo,
                        float* __restrict__ sq,
                        float* __restrict__ hp,
                        float* __restrict__ hn,
                        float* __restrict__ out) {
    int i = blockIdx.x * blockDim.x + threadIdx.x;
    if (i == 0) out[0] = 0.0f;
    if (i >= NROWS) return;

    const float* row = emb + (size_t)i * DIM;
    float s = 0.0f;
#pragma unroll
    for (int jb = 0; jb < DIM; jb += 8) {
        float4 f0 = *(const float4*)(row + jb);
        float4 f1 = *(const float4*)(row + jb + 4);
        float f[8] = {f0.x, f0.y, f0.z, f0.w, f1.x, f1.y, f1.z, f1.w};
        pack8 uh, ul;
#pragma unroll
        for (int j = 0; j < 8; ++j) {
            s += f[j] * f[j];
            bf16_t h = (bf16_t)f[j];
            uh.b[j] = h;
            ul.b[j] = (bf16_t)(f[j] - (float)h);
        }
        *(uint4*)(preHi + (size_t)i * DIM + jb) = uh.q;
        *(uint4*)(preLo + (size_t)i * DIM + jb) = ul.q;
    }
    sq[i] = s;
    hp[i] = 0.0f;                          // max of squared dist, >= 0
    hn[i] = __int_as_float(0x7F800000);    // min of squared dist, start +inf
}

// ---------- Kernel 2: WMMA Gram + batch-hard mining (squared-space) ----------
// Grid (64 row-blocks, 8 col-blocks), 256 threads = 8 wave32.
__global__ __launch_bounds__(256) void
tl_mine(const bf16_t* __restrict__ preHi,
        const bf16_t* __restrict__ preLo,
        const int*   __restrict__ labels,
        const float* __restrict__ sq,
        float* __restrict__ hp_g,
        float* __restrict__ hn_g) {
    __shared__ bf16_t sHi[2][16 * DIM];   // 2 x 4 KB, double-buffered
    __shared__ bf16_t sLo[2][16 * DIM];   // 2 x 4 KB

    const int lane = threadIdx.x & 31;
    const int wave = threadIdx.x >> 5;
    const int l15  = lane & 15;
    const int hsel = (lane >> 4) & 1;

    const int waveRowBase  = blockIdx.x * 128 + wave * 16;
    const int colBlockBase = blockIdx.y * 1024;

    // Staging geometry: thread -> (column-in-tile, 8 consecutive K), 16 B each.
    const int sc16 = threadIdx.x >> 4;
    const int skb  = (threadIdx.x & 15) * 8;
    const uint32_t ldsHi[2] = { lds_off(&sHi[0][sc16 * DIM + skb]),
                                lds_off(&sHi[1][sc16 * DIM + skb]) };
    const uint32_t ldsLo[2] = { lds_off(&sLo[0][sc16 * DIM + skb]),
                                lds_off(&sLo[1][sc16 * DIM + skb]) };

    // --- Resident A fragments (row = waveRowBase + l15), bf16 hi/lo ---
    // 16-bit A layout: element e holds K = kc*32 + (e&7) + (e>>3)*16 + hsel*8
    frag_u a_hi[4], a_lo[4];
    {
        const bf16_t* rh = preHi + (size_t)(waveRowBase + l15) * DIM;
        const bf16_t* rl = preLo + (size_t)(waveRowBase + l15) * DIM;
#pragma unroll
        for (int kc = 0; kc < 4; ++kc)
#pragma unroll
            for (int g = 0; g < 2; ++g) {
                int kb = kc * 32 + g * 16 + hsel * 8;
                a_hi[kc].q[g] = *(const uint4*)(rh + kb);
                a_lo[kc].q[g] = *(const uint4*)(rl + kb);
            }
    }

    // Accumulator VGPR r maps to row r (lanes 0-15) / r+8 (lanes 16-31).
    int   rlab[8];
    float rsq[8];
#pragma unroll
    for (int r = 0; r < 8; ++r) {
        int row  = waveRowBase + r + hsel * 8;
        rlab[r] = labels[row];
        rsq[r]  = sq[row];
    }

    const float PINF = __int_as_float(0x7F800000);
    const float NINF = __int_as_float(0xFF800000);
    float mp[8], mn[8];   // mine on m = csq - 2*dot  (d^2 = rsq + m)
#pragma unroll
    for (int r = 0; r < 8; ++r) { mp[r] = NINF; mn[r] = PINF; }

    // Prologue: async-stage chunk 0 into buffer 0.
    {
        size_t gidx = (size_t)(colBlockBase + sc16) * DIM + skb;
        async_copy_b128(ldsHi[0], preHi + gidx);
        async_copy_b128(ldsLo[0], preLo + gidx);
    }

    for (int c = 0; c < 64; ++c) {
        const int colBase = colBlockBase + c * 16;
        wait_async0();          // my stage of chunk c landed in LDS
        __syncthreads();        // everyone's stage landed; prev reads retired

        if (c + 1 < 64) {       // overlap next fill with this tile's WMMAs
            size_t gidx = (size_t)(colBase + 16 + sc16) * DIM + skb;
            int nb = (c + 1) & 1;
            async_copy_b128(ldsHi[nb], preHi + gidx);
            async_copy_b128(ldsLo[nb], preLo + gidx);
        }

        const bf16_t* bufHi = sHi[c & 1];
        const bf16_t* bufLo = sLo[c & 1];
        float csq = sq[colBase + l15];
        int   cl  = labels[colBase + l15];

        // 16x16 tile, K=128: 4 chunks x (hi*hi + hi*lo + lo*hi) WMMAs.
        v8f acc = {};
#pragma unroll
        for (int kc = 0; kc < 4; ++kc) {
            // 16-bit B layout: column N = l15; element e holds K = kc*32 + (e&7) + (e>>3)*16 + hsel*8
            frag_u bh, bl;
            const uint4* ph = (const uint4*)(bufHi + l15 * DIM + kc * 32 + hsel * 8);
            const uint4* pl = (const uint4*)(bufLo + l15 * DIM + kc * 32 + hsel * 8);
            bh.q[0] = ph[0]; bh.q[1] = ph[2];
            bl.q[0] = pl[0]; bl.q[1] = pl[2];
            acc = __builtin_amdgcn_wmma_f32_16x16x32_bf16(
                false, a_hi[kc].v, false, bh.v, (short)0, acc, false, false);
            acc = __builtin_amdgcn_wmma_f32_16x16x32_bf16(
                false, a_hi[kc].v, false, bl.v, (short)0, acc, false, false);
            acc = __builtin_amdgcn_wmma_f32_16x16x32_bf16(
                false, a_lo[kc].v, false, bh.v, (short)0, acc, false, false);
        }

        // Squared-space mining: m = csq - 2*dot; no sqrt in the hot loop.
#pragma unroll
        for (int r = 0; r < 8; ++r) {
            float m    = fmaf(-2.0f, acc[r], csq);
            bool  same = (rlab[r] == cl);
            mp[r] = fmaxf(mp[r], same ? m : NINF);
            mn[r] = fminf(mn[r], same ? PINF : m);
        }
    }

    // Reduce across the 16 lanes of each half (masks 1..8 keep halves apart).
#pragma unroll
    for (int r = 0; r < 8; ++r) {
#pragma unroll
        for (int msk = 8; msk >= 1; msk >>= 1) {
            mp[r] = fmaxf(mp[r], __shfl_xor(mp[r], msk, 32));
            mn[r] = fminf(mn[r], __shfl_xor(mn[r], msk, 32));
        }
    }
    // Clamped squared distances are >= 0 (or +inf): int atomics order correctly.
    if (l15 == 0) {
#pragma unroll
        for (int r = 0; r < 8; ++r) {
            int row   = waveRowBase + r + hsel * 8;
            float hp2 = fmaxf(0.0f, rsq[r] + mp[r]);
            float hn2 = fmaxf(0.0f, rsq[r] + mn[r]);
            atomicMax((int*)(hp_g + row), __float_as_int(hp2));
            atomicMin((int*)(hn_g + row), __float_as_int(hn2));
        }
    }
}

// ---------- Kernel 3: mean(relu(sqrt(hp2) - sqrt(hn2) + margin)) ----------
__global__ void tl_finish(const float* __restrict__ hp,
                          const float* __restrict__ hn,
                          float* __restrict__ out) {
    int i = blockIdx.x * blockDim.x + threadIdx.x;
    float v = 0.0f;
    if (i < NROWS) {
        float l = sqrtf(hp[i]) - sqrtf(hn[i]) + TL_MARGIN;
        v = l > 0.0f ? l : 0.0f;
    }
#pragma unroll
    for (int m = 16; m >= 1; m >>= 1) v += __shfl_xor(v, m, 32);
    if ((threadIdx.x & 31) == 0) atomicAdd(out, v * (1.0f / NROWS));
}

extern "C" void kernel_launch(void* const* d_in, const int* in_sizes, int n_in,
                              void* d_out, int out_size, void* d_ws, size_t ws_size,
                              hipStream_t stream) {
    const float* emb    = (const float*)d_in[0];
    const int*   labels = (const int*)d_in[1];
    float*       out    = (float*)d_out;

    // Workspace layout (16B-aligned slabs): preHi 2MB | preLo 2MB | sq | hp | hn
    bf16_t* preHi = (bf16_t*)d_ws;
    bf16_t* preLo = preHi + (size_t)NROWS * DIM;
    float*  sq    = (float*)(preLo + (size_t)NROWS * DIM);
    float*  hp    = sq + NROWS;
    float*  hn    = hp + NROWS;

    tl_prep<<<NROWS / 256, 256, 0, stream>>>(emb, preHi, preLo, sq, hp, hn, out);

    dim3 grid(64, 8);   // 64 row-blocks x 8 col-blocks
    tl_mine<<<grid, 256, 0, stream>>>(preHi, preLo, labels, sq, hp, hn);

    tl_finish<<<NROWS / 256, 256, 0, stream>>>(hp, hn, out);
}
